// ScoreLayer_15144054685953
// MI455X (gfx1250) — matrix operand order repaired and checked by reading, not compiled, not run on hardware
//
#include <hip/hip_runtime.h>

// ---------------------------------------------------------------------------
// ScoreLayer: scores[b,s,l] = left[b,s,:] @ W[l] @ right[b,s,:] + bias[l]
// B=4 S=2048 D_L=D_R=1024 L=32  -> 5.5e11 FLOP, compute-bound on bf16 WMMA.
//  - weight pre-transposed to Wt[l][e][d]: B operand contiguous in K
//  - right pre-transposed to rightT[e][pos]: fold values contiguous per lane
//    (2x b128 loads per e-tile instead of 16x u16)
//  - W streamed to LDS via CDNA5 GLOBAL_LOAD_ASYNC_TO_LDS_B128 with
//    half-chunk double buffering: DMA of half h+1 overlaps compute of half h
//  - A fragments register-resident across all e-tiles of a k-chunk
// ---------------------------------------------------------------------------

#define DL   1024
#define DR   1024
#define LQ   32
#define NPOS 8192          // B*S
#define MBLK 128           // rows per workgroup
#define KC   128           // d-chunk per LDS stage
#define NKC  (DL / KC)     // 8
#define NH   (2 * NKC)     // 16 half-chunks (512 e-rows each)
#define ROWB 272           // LDS bytes per e-row: 256 data + 16 pad (bank-safe)

typedef __attribute__((ext_vector_type(16))) __bf16 v16bf;
typedef __attribute__((ext_vector_type(8)))  float  v8f;

struct U32B { uint4 a, b; };   // 32 bytes -> one v16bf fragment

#if __has_builtin(__builtin_amdgcn_s_wait_asynccnt)
#define WAIT_ASYNC0() __builtin_amdgcn_s_wait_asynccnt(0)
#else
#define WAIT_ASYNC0() asm volatile("s_wait_asynccnt 0" ::: "memory")
#endif

static __device__ __forceinline__ unsigned short f32_to_bf16(float f) {
  unsigned u = __float_as_uint(f);
  u += 0x7fffu + ((u >> 16) & 1u);          // round-to-nearest-even
  return (unsigned short)(u >> 16);
}
static __device__ __forceinline__ float blo(unsigned w) {   // bf16 in low half
  return __uint_as_float(w << 16);
}
static __device__ __forceinline__ float bhi(unsigned w) {   // bf16 in high half
  return __uint_as_float(w & 0xffff0000u);
}

// ---- pre-pass: fp32 -> bf16 elementwise (left), vectorized -----------------
__global__ void cvt4_kernel(const float4* __restrict__ in,
                            ushort4* __restrict__ out, int n4) {
  int i = blockIdx.x * blockDim.x + threadIdx.x;
  int stride = gridDim.x * blockDim.x;
  for (; i < n4; i += stride) {
    float4 f = in[i];
    ushort4 o;
    o.x = f32_to_bf16(f.x); o.y = f32_to_bf16(f.y);
    o.z = f32_to_bf16(f.z); o.w = f32_to_bf16(f.w);
    out[i] = o;
  }
}

// ---- pre-pass: fp32 [z][R][C] -> bf16 [z][C][R] tiled transpose ------------
__global__ void trans_bf16_kernel(const float* __restrict__ in,
                                  unsigned short* __restrict__ out,
                                  int R, int C) {
  __shared__ float tile[32][33];
  const size_t base = (size_t)blockIdx.z * R * C;
  const int r0 = blockIdx.y * 32;
  const int c0 = blockIdx.x * 32;
  const int tx = threadIdx.x, ty = threadIdx.y;   // 32 x 8
  #pragma unroll
  for (int r = 0; r < 32; r += 8)
    tile[ty + r][tx] = in[base + (size_t)(r0 + ty + r) * C + c0 + tx];
  __syncthreads();
  #pragma unroll
  for (int r = 0; r < 32; r += 8)
    out[base + (size_t)(c0 + ty + r) * R + r0 + tx] = f32_to_bf16(tile[tx][ty + r]);
}

// ---- async DMA of one W half-chunk into LDS (ASYNCcnt path) ----------------
// half h: kc = h>>1, e-rows (h&1)*512..+511, 128KB -> 32 b128 per thread
static __device__ __forceinline__ void stage_half(const unsigned short* wt_l,
                                                  unsigned wlds_base,
                                                  int tid, int h) {
  const int kc = h >> 1;
  const int eh = h & 1;
  const unsigned gbase = (unsigned)(kc * (KC * 2));
  #pragma unroll 4
  for (int it = 0; it < 32; ++it) {
    int flat = it * 256 + tid;          // 16B units, 16 per e-row, 512 rows
    int e = eh * 512 + (flat >> 4);
    int u = flat & 15;
    unsigned voff  = (unsigned)e * (DL * 2) + gbase + (unsigned)(u * 16);
    unsigned laddr = wlds_base + (unsigned)e * ROWB + (unsigned)(u * 16);
    asm volatile("global_load_async_to_lds_b128 %0, %1, %2"
                 :
                 : "v"(laddr), "v"(voff), "s"(wt_l)
                 : "memory");
  }
}

// ---- main: WMMA score kernel ----------------------------------------------
// grid = (NPOS/MBLK, L), block = 256 (8 wave32)
// wave w: rows 32*(w&3)..+31 of the block, e-tiles of parity (w>>2)
__global__ void __launch_bounds__(256)
score_wmma_kernel(const unsigned short* __restrict__ leftb,
                  const unsigned short* __restrict__ rtb,   // rightT[e][pos]
                  const unsigned short* __restrict__ wtb,   // Wt[l][e][d]
                  const float* __restrict__ bias,
                  float* __restrict__ out) {
  extern __shared__ char smem[];
  unsigned short* wlds = (unsigned short*)smem;            // 1024 * 272 B
  float* slds = (float*)(smem + (size_t)DR * ROWB);        // 128 floats

  const int tid  = threadIdx.x;
  const int lane = tid & 31;
  const int wv   = tid >> 5;
  const int wrow = wv & 3;          // which 32-row slab
  const int par  = wv >> 2;         // e-tile parity
  const int l    = blockIdx.y;
  const int row0 = blockIdx.x * MBLK + wrow * 32;

  const int n  = lane & 15;         // N index within C/D fragment
  const int kh = lane >> 4;         // lane half (K grouping)

  if (tid < MBLK) slds[tid] = 0.0f;

  const unsigned short* wt_l = wtb + (size_t)l * DL * DR;
  // flat-LDS addr[31:0] is the ds offset (ISA aperture rule)
  const unsigned wlds_base = (unsigned)(size_t)(void*)wlds;

  float s0[8], s1[8];
  #pragma unroll
  for (int v = 0; v < 8; ++v) { s0[v] = 0.f; s1[v] = 0.f; }

  // pipeline prologue: half-chunk 0 in flight
  stage_half(wt_l, wlds_base, tid, 0);
  WAIT_ASYNC0();
  __syncthreads();

  v16bf a0[4], a1[4];

  for (int h = 0; h < NH; ++h) {
    const int kc = h >> 1;
    const int eh = h & 1;

    // overlap: DMA next half while computing this one (ping-pong e-halves;
    // the barrier ending iteration h-1 guarantees buffer (h+1)&1 is free)
    if (h + 1 < NH) stage_half(wt_l, wlds_base, tid, h + 1);

    // A fragments for this k-chunk (ISA 16-bit A layout: lanes 0-15
    // K={0..7,16..23}, lanes 16-31 K={8..15,24..31}), regs across e-tiles
    if (eh == 0) {
      const unsigned short* A0 =
          leftb + (size_t)(row0 + n) * DL + kc * KC + 8 * kh;
      #pragma unroll
      for (int ks = 0; ks < 4; ++ks) {
        const unsigned short* p0 = A0 + ks * 32;
        U32B t0; t0.a = *(const uint4*)(p0); t0.b = *(const uint4*)(p0 + 16);
        a0[ks] = __builtin_bit_cast(v16bf, t0);
        const unsigned short* p1 = p0 + 16 * DL;
        U32B t1; t1.a = *(const uint4*)(p1); t1.b = *(const uint4*)(p1 + 16);
        a1[ks] = __builtin_bit_cast(v16bf, t1);
      }
    }

    #pragma unroll 2
    for (int t = 0; t < 16; ++t) {
      const int e0 = eh * 512 + (2 * t + par) * 16;
      const char* brow = (const char*)wlds + (size_t)(e0 + n) * ROWB;
      v8f acc0 = {}; v8f acc1 = {};
      #pragma unroll
      for (int ks = 0; ks < 4; ++ks) {
        // B layout: lane n holds column e0+n, K={8kh..8kh+7, 16+8kh..16+8kh+7}
        const char* bp = brow + ks * 64 + kh * 16;
        U32B tb; tb.a = *(const uint4*)bp; tb.b = *(const uint4*)(bp + 32);
        v16bf bf = __builtin_bit_cast(v16bf, tb);
        acc0 = __builtin_amdgcn_wmma_f32_16x16x32_bf16(
            false, a0[ks], false, bf, (short)0, acc0, false, false);
        acc1 = __builtin_amdgcn_wmma_f32_16x16x32_bf16(
            false, a1[ks], false, bf, (short)0, acc1, false, false);
      }
      // fold Y against rightT: lane's 8 fold values are 16B contiguous
      const unsigned short* rq =
          rtb + ((size_t)(e0 + n) * NPOS + row0 + 8 * kh);
      uint4 ra = *(const uint4*)(rq);        // rows row0+8kh+0..7   -> s0
      uint4 rb = *(const uint4*)(rq + 16);   // rows row0+16+8kh+0..7 -> s1
      s0[0] += acc0[0] * blo(ra.x); s0[1] += acc0[1] * bhi(ra.x);
      s0[2] += acc0[2] * blo(ra.y); s0[3] += acc0[3] * bhi(ra.y);
      s0[4] += acc0[4] * blo(ra.z); s0[5] += acc0[5] * bhi(ra.z);
      s0[6] += acc0[6] * blo(ra.w); s0[7] += acc0[7] * bhi(ra.w);
      s1[0] += acc1[0] * blo(rb.x); s1[1] += acc1[1] * bhi(rb.x);
      s1[2] += acc1[2] * blo(rb.y); s1[3] += acc1[3] * bhi(rb.y);
      s1[4] += acc1[4] * blo(rb.z); s1[5] += acc1[5] * bhi(rb.z);
      s1[6] += acc1[6] * blo(rb.w); s1[7] += acc1[7] * bhi(rb.w);
    }

    if (h + 1 < NH) WAIT_ASYNC0();   // next half landed (overlapped w/ compute)
    __syncthreads();                 // publish; also frees buffer for h+2
  }

  // reduce over the 16 N-lanes of each half, combine parity waves via LDS
  #pragma unroll
  for (int v = 0; v < 8; ++v) {
    float x0 = s0[v], x1 = s1[v];
    #pragma unroll
    for (int m = 1; m < 16; m <<= 1) {
      x0 += __shfl_xor(x0, m, 32);
      x1 += __shfl_xor(x1, m, 32);
    }
    if (n == 0) {
      atomicAdd(&slds[wrow * 32 +      8 * kh + v], x0);
      atomicAdd(&slds[wrow * 32 + 16 + 8 * kh + v], x1);
    }
  }
  __syncthreads();
  if (tid < MBLK) {
    int grow = blockIdx.x * MBLK + tid;
    out[(size_t)grow * LQ + l] = slds[tid] + bias[l];
  }
}

// ---------------------------------------------------------------------------
extern "C" void kernel_launch(void* const* d_in, const int* in_sizes, int n_in,
                              void* d_out, int out_size, void* d_ws, size_t ws_size,
                              hipStream_t stream) {
  (void)in_sizes; (void)n_in; (void)out_size; (void)ws_size;
  const float* left  = (const float*)d_in[0];
  const float* right = (const float*)d_in[1];
  const float* w     = (const float*)d_in[2];
  const float* bias  = (const float*)d_in[3];
  float* out = (float*)d_out;

  // workspace (bf16): left 16MB | rightT 16MB | Wt 64MB
  unsigned short* leftb = (unsigned short*)d_ws;
  unsigned short* rtb   = leftb + (size_t)NPOS * DL;
  unsigned short* wtb   = rtb   + (size_t)NPOS * DR;

  cvt4_kernel<<<2048, 256, 0, stream>>>((const float4*)left,
                                        (ushort4*)leftb, NPOS * DL / 4);
  // rightT[e][pos] = bf16(right[pos][e])
  trans_bf16_kernel<<<dim3(DR / 32, NPOS / 32, 1), dim3(32, 8), 0, stream>>>(
      right, rtb, NPOS, DR);
  // Wt[l][e][d] = bf16(W[l][d][e])
  trans_bf16_kernel<<<dim3(DR / 32, DL / 32, LQ), dim3(32, 8), 0, stream>>>(
      w, wtb, DL, DR);

  size_t shmem = (size_t)DR * ROWB + MBLK * sizeof(float); // 278528 + 512 B
  score_wmma_kernel<<<dim3(NPOS / MBLK, LQ), 256, shmem, stream>>>(
      leftb, rtb, wtb, bias, out);
}
